// DecoderAttention_7241314861844
// MI455X (gfx1250) — matrix operand order repaired and checked by reading, not compiled, not run on hardware
//
#include <hip/hip_runtime.h>
#include <hip/hip_bf16.h>
#include <stdint.h>

typedef _Float16 half4 __attribute__((ext_vector_type(4)));
typedef _Float16 half8 __attribute__((ext_vector_type(8)));
typedef _Float16 v16h  __attribute__((ext_vector_type(16)));
typedef float    v8f   __attribute__((ext_vector_type(8)));

#define N_SEQ   2048
#define M_SEQ   1024
#define NK_SEQ  3072
#define D_MODEL 1024
#define N_HEADS 16
#define DH      64
#define LOG2E   1.4426950408889634f
#define MASK2   (-72134.75f)   /* -50000 * log2(e) */

__device__ __forceinline__ v16h join16(half8 lo, half8 hi) {
  return __builtin_shufflevector(lo, hi, 0,1,2,3,4,5,6,7,8,9,10,11,12,13,14,15);
}

__device__ __forceinline__ v8f zero8() {
  v8f z = {0.f,0.f,0.f,0.f,0.f,0.f,0.f,0.f};
  return z;
}

// A-matrix 16x32 f16 fragment (ISA 7.12.2): row = lane&15; halves 0..7 hold
// K = kb..kb+7, halves 8..15 hold K = 16+kb..16+kb+7 with kb = (lane>>4)*8.
// base -> (row0, k0) of row-major storage, `stride` halves per row.
__device__ __forceinline__ v16h load_afrag(const _Float16* base, int stride, int lane) {
  const _Float16* p = base + (lane & 15) * stride + ((lane >> 4) << 3);
  half8 lo = *(const half8*)(p);
  half8 hi = *(const half8*)(p + 16);
  return join16(lo, hi);
}

// B-matrix 32x16 f16 fragment: col = lane&15; halves 0..7 hold
// K = kb..kb+7, halves 8..15 hold K = 16+kb..16+kb+7 with kb = (lane>>4)*8.
// base -> (k0, col0) of column-major storage (`stride` halves per column).
__device__ __forceinline__ v16h load_bfrag(const _Float16* base, int stride, int lane) {
  const _Float16* p = base + (lane & 15) * stride + ((lane >> 4) << 3);
  half8 lo = *(const half8*)(p);
  half8 hi = *(const half8*)(p + 16);
  return join16(lo, hi);
}

#define WMMA_F16(A, B, C) \
  __builtin_amdgcn_wmma_f32_16x16x32_f16(false, (A), false, (B), (short)0, (C), false, false)

// ---------------------------------------------------------------- converts
__global__ __launch_bounds__(256)
void cvt_f32_to_f16(const float* __restrict__ src, _Float16* __restrict__ dst, int n4) {
  int i = blockIdx.x * 256 + threadIdx.x;
  if (i < n4) {
    float4 v = ((const float4*)src)[i];
    half4 h = { (_Float16)v.x, (_Float16)v.y, (_Float16)v.z, (_Float16)v.w };
    ((half4*)dst)[i] = h;
  }
}

// ---------------------------------------------------------------- GEMM
// C[M,N] = (A[M,K] @ B[K,N] + bias) * alpha ; B pre-converted f16.
// MODE 0: A=f32 (A0), out f16 O16a (Q projection, alpha folds scale*log2e)
// MODE 1: A=f32 rows from concat(x, context) per batch, split cols -> K/V f16
// MODE 2: A=f16 (A16), out f32 (final projection)
#define GTM 128
#define GTN 128
#define GTK 32
#define LPAD 8
#define KSTR (GTK + LPAD)

template<int MODE>
__global__ __launch_bounds__(256)
void gemm_wmma(const float* __restrict__ A0, const float* __restrict__ A1,
               const _Float16* __restrict__ A16, const _Float16* __restrict__ B16,
               const float* __restrict__ bias,
               _Float16* __restrict__ O16a, _Float16* __restrict__ O16b,
               float* __restrict__ O32,
               int M, int N, int K, float alpha)
{
  __shared__ __align__(16) _Float16 As[2][GTM][KSTR];   // row-major A tiles
  __shared__ __align__(16) _Float16 Bs[2][GTN][KSTR];   // column-major B tiles

  const int tid  = threadIdx.x;
  const int lane = tid & 31;
  const int wid  = tid >> 5;
  const int m0 = blockIdx.y * GTM;
  const int n0 = blockIdx.x * GTN;
  const int wm = (wid & 3) * 32;   // 4 waves along M
  const int wn = (wid >> 2) * 64;  // 2 waves along N

  v8f acc[2][4];
#pragma unroll
  for (int i = 0; i < 2; ++i)
#pragma unroll
    for (int j = 0; j < 4; ++j) acc[i][j] = zero8();

  // staging registers (prefetch of next k-tile)
  float4 aF[4];
  half8  aH[2];
  half8  bH[2];

  auto load_regs = [&](int k0) {
    if constexpr (MODE == 2) {
#pragma unroll
      for (int p = 0; p < 2; ++p) {
        int g = tid + p * 256, r = g >> 2, c8 = (g & 3) << 3;
        aH[p] = *(const half8*)(A16 + (size_t)(m0 + r) * K + k0 + c8);
      }
    } else {
#pragma unroll
      for (int p = 0; p < 4; ++p) {
        int g = tid + p * 256, r = g >> 3, c4 = (g & 7) << 2;
        int row = m0 + r;
        const float* ap;
        if constexpr (MODE == 1) {
          int bb = row / NK_SEQ;
          int s  = row - bb * NK_SEQ;
          ap = (s < N_SEQ) ? (A0 + ((size_t)bb * N_SEQ + s) * (size_t)K)
                           : (A1 + ((size_t)bb * M_SEQ + (s - N_SEQ)) * (size_t)K);
        } else {
          ap = A0 + (size_t)row * K;
        }
        aF[p] = *(const float4*)(ap + k0 + c4);
      }
    }
#pragma unroll
    for (int p = 0; p < 2; ++p) {
      int g = tid + p * 256, kr = g >> 4, c8 = (g & 15) << 3;
      bH[p] = *(const half8*)(B16 + (size_t)(k0 + kr) * N + n0 + c8);
    }
  };

  auto store_regs = [&](int buf) {
    if constexpr (MODE == 2) {
#pragma unroll
      for (int p = 0; p < 2; ++p) {
        int g = tid + p * 256, r = g >> 2, c8 = (g & 3) << 3;
        *(half8*)&As[buf][r][c8] = aH[p];
      }
    } else {
#pragma unroll
      for (int p = 0; p < 4; ++p) {
        int g = tid + p * 256, r = g >> 3, c4 = (g & 7) << 2;
        half4 h = { (_Float16)aF[p].x, (_Float16)aF[p].y,
                    (_Float16)aF[p].z, (_Float16)aF[p].w };
        *(half4*)&As[buf][r][c4] = h;
      }
    }
#pragma unroll
    for (int p = 0; p < 2; ++p) {
      int g = tid + p * 256, kr = g >> 4, c8 = (g & 15) << 3;
#pragma unroll
      for (int q = 0; q < 8; ++q) Bs[buf][c8 + q][kr] = bH[p][q];
    }
  };

  const int KT = K >> 5;
  load_regs(0);
  store_regs(0);
  __syncthreads();

  for (int kt = 0; kt < KT; ++kt) {
    const int  cur  = kt & 1;
    const bool more = (kt + 1) < KT;
    if (more) load_regs((kt + 1) << 5);   // global loads overlap WMMA below

    v16h af[2], bf[4];
#pragma unroll
    for (int i = 0; i < 2; ++i) af[i] = load_afrag(&As[cur][wm + i * 16][0], KSTR, lane);
#pragma unroll
    for (int j = 0; j < 4; ++j) bf[j] = load_bfrag(&Bs[cur][wn + j * 16][0], KSTR, lane);
#pragma unroll
    for (int i = 0; i < 2; ++i)
#pragma unroll
      for (int j = 0; j < 4; ++j)
        acc[i][j] = WMMA_F16(af[i], bf[j], acc[i][j]);

    if (more) store_regs(cur ^ 1);
    __syncthreads();
  }

  // ---- epilogue (C layout: col = lane&15 ; row = v + 8*(lane>=16))
  const int cl = lane & 15;
  const int rh = (lane >> 4) << 3;
#pragma unroll
  for (int i = 0; i < 2; ++i) {
#pragma unroll
    for (int j = 0; j < 4; ++j) {
      int col = n0 + wn + j * 16 + cl;
      float bv = bias[col];
#pragma unroll
      for (int v = 0; v < 8; ++v) {
        int row = m0 + wm + i * 16 + rh + v;
        float val = (acc[i][j][v] + bv) * alpha;
        if constexpr (MODE == 0) {
          O16a[(size_t)row * N + col] = (_Float16)val;
        } else if constexpr (MODE == 1) {
          if (col < D_MODEL) O16a[(size_t)row * D_MODEL + col] = (_Float16)val;
          else               O16b[(size_t)row * D_MODEL + (col - D_MODEL)] = (_Float16)val;
        } else {
          O32[(size_t)row * N + col] = val;
        }
      }
    }
  }
}

// ---------------------------------------------------------------- attention
// Flash attention over 3072 keys (2048 self + 1024 cross), causal on self.
// Q pre-scaled by dh^-0.5 * log2(e); softmax via exp2.
// Block: 128 threads (4 waves), 128 q-rows; wave owns 32 rows.
// K tiles staged with gfx1250 async global->LDS DMA (ASYNCcnt), V tiles
// register-prefetched and transposed; both double-buffered.
__global__ __launch_bounds__(128)
void attn_flash(const _Float16* __restrict__ Qh, const _Float16* __restrict__ Kh,
                const _Float16* __restrict__ Vh,
                const unsigned char* __restrict__ qmask,
                const unsigned char* __restrict__ cmask,
                _Float16* __restrict__ Ah)
{
  __shared__ __align__(16) _Float16 Ks[2][32][DH + 8];     // [key][dh]
  __shared__ __align__(16) _Float16 Vs[2][DH][32 + 8];     // [dh][key]
  __shared__ __align__(16) _Float16 Pb[4][32][32 + 8];     // per-wave P tile

  const int tid  = threadIdx.x;
  const int lane = tid & 31;
  const int wid  = tid >> 5;

  const int qt = blockIdx.x & 15;
  const int bh = blockIdx.x >> 4;
  const int h  = bh & (N_HEADS - 1);
  const int bb = bh >> 4;

  const int q0 = qt * 128;
  const int qw = q0 + wid * 32;
  const int rh = (lane >> 4) << 3;

  const int nself  = (q0 + 128) >> 5;           // self key tiles (causal skip)
  const int ntiles = nself + (M_SEQ >> 5);

  auto j_of = [&](int jt) {
    return (jt < nself) ? (jt << 5) : (N_SEQ + ((jt - nself) << 5));
  };

  // async K tile stage: each thread copies 2 x 16B segments global -> LDS
  auto issue_k_async = [&](int j0, int buf) {
    uint64_t sbase = (uint64_t)(Kh + ((size_t)(bb * NK_SEQ + j0)) * D_MODEL + h * DH);
#pragma unroll
    for (int p = 0; p < 2; ++p) {
      int g = tid + p * 128, r = g >> 3, c8 = (g & 7) << 3;
      uint32_t voff = (uint32_t)((r * D_MODEL + c8) * sizeof(_Float16));
      uint32_t loff = (uint32_t)(uintptr_t)(&Ks[buf][r][c8]);
      asm volatile("global_load_async_to_lds_b128 %0, %1, %2"
                   :: "v"(loff), "v"(voff), "s"(sbase) : "memory");
    }
  };
  auto load_v_regs = [&](int j0, half8* vr) {
#pragma unroll
    for (int p = 0; p < 2; ++p) {
      int g = tid + p * 128, r = g >> 3, c8 = (g & 7) << 3;
      vr[p] = *(const half8*)(Vh + ((size_t)(bb * NK_SEQ + j0 + r)) * D_MODEL + h * DH + c8);
    }
  };
  auto store_v = [&](int buf, const half8* vr) {
#pragma unroll
    for (int p = 0; p < 2; ++p) {
      int g = tid + p * 128, r = g >> 3, c8 = (g & 7) << 3;
#pragma unroll
      for (int q = 0; q < 8; ++q) Vs[buf][c8 + q][r] = vr[p][q];
    }
  };

  // Q fragments [row-tile][k-step], held in registers for the whole kernel
  v16h qf[2][2];
#pragma unroll
  for (int rt = 0; rt < 2; ++rt)
#pragma unroll
    for (int kk = 0; kk < 2; ++kk)
      qf[rt][kk] = load_afrag(
          Qh + ((size_t)(bb * N_SEQ + qw + rt * 16)) * D_MODEL + h * DH + kk * 32,
          D_MODEL, lane);

  unsigned qok[2];
#pragma unroll
  for (int rt = 0; rt < 2; ++rt) {
    unsigned b = 0;
#pragma unroll
    for (int v = 0; v < 8; ++v) {
      int qr = qw + rt * 16 + rh + v;
      b |= (qmask[bb * N_SEQ + qr] ? 1u : 0u) << v;
    }
    qok[rt] = b;
  }

  float mrow[2][8], lrow[2][8];
  v8f o[2][4];
#pragma unroll
  for (int rt = 0; rt < 2; ++rt) {
#pragma unroll
    for (int v = 0; v < 8; ++v) { mrow[rt][v] = -1e30f; lrow[rt][v] = 0.f; }
#pragma unroll
    for (int t = 0; t < 4; ++t) o[rt][t] = zero8();
  }

  // prologue: stage tile 0 into buffer 0
  half8 vreg[2];
  issue_k_async(j_of(0), 0);
  load_v_regs(j_of(0), vreg);
  store_v(0, vreg);
  asm volatile("s_wait_asynccnt 0x0" ::: "memory");
  __syncthreads();

  for (int jt = 0; jt < ntiles; ++jt) {
    const int  cur  = jt & 1;
    const bool more = (jt + 1) < ntiles;
    const int  j0   = j_of(jt);

    if (more) {                       // prefetch next tile (overlaps compute)
      issue_k_async(j_of(jt + 1), cur ^ 1);
      load_v_regs(j_of(jt + 1), vreg);
    }

    // ---- S = Q @ K^T  (log2 domain)
    v8f s[2][2];
    s[0][0] = zero8(); s[0][1] = zero8(); s[1][0] = zero8(); s[1][1] = zero8();
#pragma unroll
    for (int kk = 0; kk < 2; ++kk) {
      v16h bk0 = load_bfrag(&Ks[cur][ 0][kk * 32], DH + 8, lane);
      v16h bk1 = load_bfrag(&Ks[cur][16][kk * 32], DH + 8, lane);
      s[0][0] = WMMA_F16(qf[0][kk], bk0, s[0][0]);
      s[0][1] = WMMA_F16(qf[0][kk], bk1, s[0][1]);
      s[1][0] = WMMA_F16(qf[1][kk], bk0, s[1][0]);
      s[1][1] = WMMA_F16(qf[1][kk], bk1, s[1][1]);
    }

    // ---- mask + online softmax
    const int j_a = j0 + (lane & 15);
    const int j_b = j_a + 16;
    const bool kok_a = (j_a < N_SEQ) ? (qmask[bb * N_SEQ + j_a] != 0)
                                     : (cmask[bb * M_SEQ + (j_a - N_SEQ)] != 0);
    const bool kok_b = (j_b < N_SEQ) ? (qmask[bb * N_SEQ + j_b] != 0)
                                     : (cmask[bb * M_SEQ + (j_b - N_SEQ)] != 0);
#pragma unroll
    for (int rt = 0; rt < 2; ++rt) {
#pragma unroll
      for (int v = 0; v < 8; ++v) {
        const int qr = qw + rt * 16 + rh + v;
        const bool qo = (qok[rt] >> v) & 1;
        float s0 = s[rt][0][v];
        float s1 = s[rt][1][v];
        bool a0 = qo && kok_a && !((j_a < N_SEQ) && (j_a > qr));
        bool a1 = qo && kok_b && !((j_b < N_SEQ) && (j_b > qr));
        s0 = a0 ? s0 : MASK2;
        s1 = a1 ? s1 : MASK2;
        float mv = fmaxf(s0, s1);
        mv = fmaxf(mv, __shfl_xor(mv, 1, 32));
        mv = fmaxf(mv, __shfl_xor(mv, 2, 32));
        mv = fmaxf(mv, __shfl_xor(mv, 4, 32));
        mv = fmaxf(mv, __shfl_xor(mv, 8, 32));
        float mn   = fmaxf(mrow[rt][v], mv);
        float corr = exp2f(mrow[rt][v] - mn);
        mrow[rt][v] = mn;
        float p0 = exp2f(s0 - mn);
        float p1 = exp2f(s1 - mn);
        lrow[rt][v] = lrow[rt][v] * corr + p0 + p1;
#pragma unroll
        for (int t = 0; t < 4; ++t) o[rt][t][v] *= corr;
        int prow = rt * 16 + rh + v;
        Pb[wid][prow][(lane & 15)]      = (_Float16)p0;
        Pb[wid][prow][16 + (lane & 15)] = (_Float16)p1;
      }
    }

    // ---- O += P @ V  (P re-read from per-wave LDS as A-fragments)
    v16h pa0 = load_afrag(&Pb[wid][ 0][0], 32 + 8, lane);
    v16h pa1 = load_afrag(&Pb[wid][16][0], 32 + 8, lane);
#pragma unroll
    for (int t = 0; t < 4; ++t) {
      v16h vf = load_bfrag(&Vs[cur][t * 16][0], 32 + 8, lane);
      o[0][t] = WMMA_F16(pa0, vf, o[0][t]);
      o[1][t] = WMMA_F16(pa1, vf, o[1][t]);
    }

    if (more) store_v(cur ^ 1, vreg);
    asm volatile("s_wait_asynccnt 0x0" ::: "memory");
    __syncthreads();
  }

  // ---- finalize: reduce l across lanes, normalize, store f16
#pragma unroll
  for (int rt = 0; rt < 2; ++rt) {
#pragma unroll
    for (int v = 0; v < 8; ++v) {
      float lv = lrow[rt][v];
      lv += __shfl_xor(lv, 1, 32);
      lv += __shfl_xor(lv, 2, 32);
      lv += __shfl_xor(lv, 4, 32);
      lv += __shfl_xor(lv, 8, 32);
      float rinv = 1.0f / lv;
      int qr = qw + rt * 16 + rh + v;
      size_t base = ((size_t)(bb * N_SEQ + qr)) * D_MODEL + h * DH;
#pragma unroll
      for (int t = 0; t < 4; ++t)
        Ah[base + t * 16 + (lane & 15)] = (_Float16)(o[rt][t][v] * rinv);
    }
  }
}

// ---------------------------------------------------------------- launch
extern "C" void kernel_launch(void* const* d_in, const int* in_sizes, int n_in,
                              void* d_out, int out_size, void* d_ws, size_t ws_size,
                              hipStream_t stream) {
  const float* x    = (const float*)d_in[0];
  const float* ctx  = (const float*)d_in[1];
  const float* Wq   = (const float*)d_in[2];
  const float* bq   = (const float*)d_in[3];
  const float* Wkv  = (const float*)d_in[4];
  const float* bkv  = (const float*)d_in[5];
  const float* Wo   = (const float*)d_in[6];
  const float* bo   = (const float*)d_in[7];
  const unsigned char* mask  = (const unsigned char*)d_in[8];
  const unsigned char* cmask = (const unsigned char*)d_in[9];
  float* out = (float*)d_out;

  _Float16* ws    = (_Float16*)d_ws;
  _Float16* Wq_h  = ws;                                  // 1024*1024
  _Float16* Wkv_h = Wq_h  + (size_t)1024 * 1024;         // 1024*2048
  _Float16* Wo_h  = Wkv_h + (size_t)1024 * 2048;         // 1024*1024
  _Float16* Qhp   = Wo_h  + (size_t)1024 * 1024;         // 4096*1024
  _Float16* Khp   = Qhp   + (size_t)4096 * 1024;         // 6144*1024
  _Float16* Vhp   = Khp   + (size_t)6144 * 1024;         // 6144*1024
  _Float16* Ahp   = Vhp   + (size_t)6144 * 1024;         // 4096*1024

  // 1) weight conversion f32 -> f16
  cvt_f32_to_f16<<<(1024 * 1024 / 4) / 256, 256, 0, stream>>>(Wq,  Wq_h,  1024 * 1024 / 4);
  cvt_f32_to_f16<<<(1024 * 2048 / 4) / 256, 256, 0, stream>>>(Wkv, Wkv_h, 1024 * 2048 / 4);
  cvt_f32_to_f16<<<(1024 * 1024 / 4) / 256, 256, 0, stream>>>(Wo,  Wo_h,  1024 * 1024 / 4);

  // 2) Q = (x@Wq + bq) * dh^-0.5 * log2(e)
  gemm_wmma<0><<<dim3(1024 / GTN, 4096 / GTM), 256, 0, stream>>>(
      x, nullptr, nullptr, Wq_h, bq, Qhp, nullptr, nullptr,
      4096, 1024, 1024, 0.125f * LOG2E);

  // 3) K,V = concat(x,ctx)@Wkv + bkv
  gemm_wmma<1><<<dim3(2048 / GTN, 6144 / GTM), 256, 0, stream>>>(
      x, ctx, nullptr, Wkv_h, bkv, Khp, Vhp, nullptr,
      6144, 2048, 1024, 1.0f);

  // 4) flash attention -> Ah (f16, [b, n, h*dh] layout)
  attn_flash<<<dim3(2 * N_HEADS * (N_SEQ / 128)), 128, 0, stream>>>(
      Qhp, Khp, Vhp, mask, cmask, Ahp);

  // 5) out = Ah @ Wo + bo (f32)
  gemm_wmma<2><<<dim3(1024 / GTN, 4096 / GTM), 256, 0, stream>>>(
      nullptr, nullptr, Ahp, Wo_h, bo, nullptr, nullptr, out,
      4096, 1024, 1024, 1.0f);
}
